// GCNWithMultiHeadGATAndTCN_69312182223586
// MI455X (gfx1250) — matrix operand order repaired and checked by reading, not compiled, not run on hardware
//
#include <hip/hip_runtime.h>
#include <hip/hip_bf16.h>

// ---------------------------------------------------------------------------
// GCN + multi-head GAT + TCN for MI455X (gfx1250). All contractions on
// v_wmma_f32_16x16x32_f16 (f16 operands, f32 accumulate). Wave32, blockDim=32
// (EXEC all ones for WMMA). GEMMs register-block 32x64 per wave:
// per K-step 2 A-fragments + 4 B-fragments feed 8 WMMAs (2.7x the
// arithmetic intensity of 1-tile waves).
// ---------------------------------------------------------------------------

typedef __attribute__((ext_vector_type(16))) _Float16 v16h;
typedef __attribute__((ext_vector_type(8)))  _Float16 v8h;
typedef __attribute__((ext_vector_type(8)))  float    v8f;

#define B_   16
#define N_   1024
#define NF_  128
#define NH_  256
#define NO_  256
#define H_   4
#define DH_  64
#define EPS_ 1e-5f
#define ALPHA_ 0.2f

__device__ __forceinline__ v8f wmma_f16(v16h a, v16h b, v8f c) {
  return __builtin_amdgcn_wmma_f32_16x16x32_f16(false, a, false, b, (short)0, c,
                                                false, false);
}

// A fragment: 16x32 f16 tile, row-major source. Lane (r = lane&15, hf=lane>>4)
// holds A[row0+r][k0 + 16*hf + 0..15] (contiguous).
__device__ __forceinline__ v16h load_a16(const _Float16* A, int lda, int row0,
                                         int k0) {
  int lane = threadIdx.x & 31;
  int r = lane & 15, hf = lane >> 4;
  const _Float16* p = A + (size_t)(row0 + r) * lda + k0 + 16 * hf;
  v8h lo = *(const v8h*)p;
  v8h hi = *(const v8h*)(p + 8);
  v16h a;
#pragma unroll
  for (int i = 0; i < 8; ++i) { a[i] = lo[i]; a[i + 8] = hi[i]; }
  return a;
}

// B fragment from COLUMN-MAJOR storage Bc[col*ldb + k]. Lane (c=lane&15,
// hf=lane>>4) holds B[k0 + 16*hf + j][col0+c] for j = 0..15 (contiguous).
__device__ __forceinline__ v16h load_b16(const _Float16* Bc, int ldb, int col0,
                                         int k0) {
  int lane = threadIdx.x & 31;
  int c = lane & 15, hf = lane >> 4;
  const _Float16* p = Bc + (size_t)(col0 + c) * ldb + k0 + 16 * hf;
  v8h lo = *(const v8h*)p;
  v8h hi = *(const v8h*)(p + 8);
  v16h b;
#pragma unroll
  for (int i = 0; i < 8; ++i) { b[i] = lo[i]; b[i + 8] = hi[i]; }
  return b;
}

// ----------------------------- pack kernels --------------------------------

__global__ void pack_f32_to_f16(const float* src, _Float16* dst, int n) {
  int i = blockIdx.x * 256 + threadIdx.x;
  if (i < n) dst[i] = (_Float16)src[i];
}

// gc_w (128x256) -> [o][f] f16 (column-major for GEMM B)
__global__ void pack_gcw(const float* w, _Float16* out) {
  int i = blockIdx.x * 256 + threadIdx.x;  // 256*128
  int o = i >> 7, f = i & 127;
  out[i] = (_Float16)w[f * NH_ + o];
}

// gat_w (4,256,64) -> [h][d][f] f16
__global__ void pack_gatw(const float* w, _Float16* out) {
  int i = blockIdx.x * 256 + threadIdx.x;  // 4*64*256
  int h = i >> 14;
  int r = i & 16383;
  int d = r >> 8, f = r & 255;
  out[i] = (_Float16)w[h * (NH_ * DH_) + f * DH_ + d];
}

// conv_w (256,256,3) -> [k][o][i] f16
__global__ void pack_convw(const float* w, _Float16* out) {
  int i = blockIdx.x * 256 + threadIdx.x;  // 3*256*256
  int k = i >> 16;
  int r = i & 65535;
  int o = r >> 8, ic = r & 255;
  out[i] = (_Float16)w[o * (NO_ * 3) + ic * 3 + k];
}

// ------------------------------- GEMM 1 ------------------------------------
// h0 = x @ gc_w ; stored TRANSPOSED per batch: h0t[b][o][m]  (f16)
// 32x64 per wave: grid = (16384/32) * (256/64) = 2048
__global__ __launch_bounds__(32) void gemm1_kernel(const _Float16* xh,
                                                   const _Float16* gcwT,
                                                   _Float16* h0t) {
  int tile = blockIdx.x;
  int mg = tile >> 2, og = tile & 3;
  int m0 = mg * 32, o0 = og * 64;
  v8f acc[2][4] = {};
#pragma unroll
  for (int k0 = 0; k0 < NF_; k0 += 32) {
    v16h a0 = load_a16(xh, NF_, m0, k0);
    v16h a1 = load_a16(xh, NF_, m0 + 16, k0);
#pragma unroll
    for (int c = 0; c < 4; ++c) {
      v16h b = load_b16(gcwT, NF_, o0 + 16 * c, k0);
      acc[0][c] = wmma_f16(a0, b, acc[0][c]);
      acc[1][c] = wmma_f16(a1, b, acc[1][c]);
    }
  }
  int lane = threadIdx.x & 31;
  int hf = lane >> 4;
  int bb = m0 >> 10;  // 32-row group never crosses a batch boundary
#pragma unroll
  for (int rt = 0; rt < 2; ++rt) {
    int ml = (m0 & 1023) + rt * 16 + 8 * hf;
#pragma unroll
    for (int c = 0; c < 4; ++c) {
      int o = o0 + 16 * c + (lane & 15);
      _Float16* dst = h0t + (((size_t)bb << 8) + o) * N_ + ml;
      v8h ov;
#pragma unroll
      for (int i = 0; i < 8; ++i) ov[i] = (_Float16)acc[rt][c][i];
      *(v8h*)dst = ov;
    }
  }
}

// ------------------------------- GEMM 2 ------------------------------------
// h1[b][n][o] = relu( adj @ h0[b] + gc_b )   (f32 out, for BN stats)
// 32x64 per wave: grid = 16 * (1024/32) * (256/64) = 2048
__global__ __launch_bounds__(32) void gemm2_kernel(const _Float16* adjh,
                                                   const _Float16* h0t,
                                                   const float* gc_b,
                                                   float* h1) {
  int tile = blockIdx.x;
  int b = tile >> 7;
  int rem = tile & 127;
  int ng = rem >> 2, og = rem & 3;
  int n0 = ng * 32, o0 = og * 64;
  const _Float16* Bc = h0t + ((size_t)b << 8) * N_;  // [o][m]
  v8f acc[2][4] = {};
  for (int k0 = 0; k0 < N_; k0 += 32) {
    v16h a0 = load_a16(adjh, N_, n0, k0);
    v16h a1 = load_a16(adjh, N_, n0 + 16, k0);
#pragma unroll
    for (int c = 0; c < 4; ++c) {
      v16h bm = load_b16(Bc, N_, o0 + 16 * c, k0);
      acc[0][c] = wmma_f16(a0, bm, acc[0][c]);
      acc[1][c] = wmma_f16(a1, bm, acc[1][c]);
    }
  }
  int lane = threadIdx.x & 31;
  int hf = lane >> 4;
#pragma unroll
  for (int c = 0; c < 4; ++c) {
    int o = o0 + 16 * c + (lane & 15);
    float bias = gc_b[o];
#pragma unroll
    for (int rt = 0; rt < 2; ++rt) {
      float* dst = h1 + ((size_t)b * N_ + n0 + rt * 16 + 8 * hf) * NH_ + o;
#pragma unroll
      for (int i = 0; i < 8; ++i)
        dst[(size_t)i * NH_] = fmaxf(acc[rt][c][i] + bias, 0.f);
    }
  }
}

// --------------------------- BatchNorm stats -------------------------------
__global__ void bn_stats_kernel(const float* X, int R, int C,
                                const float* gamma, const float* beta,
                                float* scale, float* shift) {
  int c = blockIdx.x;
  int t = threadIdx.x;
  float s = 0.f, s2 = 0.f;
  for (int r = t; r < R; r += 256) {
    float v = X[(size_t)r * C + c];
    s += v;
    s2 += v * v;
  }
  __shared__ float sh[256], sh2[256];
  sh[t] = s;
  sh2[t] = s2;
  __syncthreads();
  for (int off = 128; off > 0; off >>= 1) {
    if (t < off) { sh[t] += sh[t + off]; sh2[t] += sh2[t + off]; }
    __syncthreads();
  }
  if (t == 0) {
    float mu = sh[0] / (float)R;
    float var = sh2[0] / (float)R - mu * mu;
    float sc = gamma[c] * rsqrtf(var + EPS_);
    scale[c] = sc;
    shift[c] = beta[c] - mu * sc;
  }
}

// BN apply + relu + quantize f16 (for the GAT GEMM)
__global__ void bn1_apply_kernel(const float* h1, const float* scale,
                                 const float* shift, _Float16* h2h) {
  size_t i = (size_t)blockIdx.x * 256 + threadIdx.x;
  int c = (int)(i & 255);
  float v = h1[i] * scale[c] + shift[c];
  h2h[i] = (_Float16)fmaxf(v, 0.f);
}

// ------------------------------- GEMM 3 ------------------------------------
// hp[b][h][n][d] = h2 @ gat_w[h]; also stores hpT[b][h][d][n]
// 32 rows x all 64 d-cols per wave: grid = (16384/32) * 4 heads = 2048
__global__ __launch_bounds__(32) void gemm3_kernel(const _Float16* h2h,
                                                   const _Float16* gatwT,
                                                   _Float16* hp,
                                                   _Float16* hpT) {
  int tile = blockIdx.x;
  int head = tile & 3;
  int mg = tile >> 2;
  int m0 = mg * 32;
  const _Float16* Bc = gatwT + (size_t)head * DH_ * NH_;
  v8f acc[2][4] = {};
#pragma unroll
  for (int k0 = 0; k0 < NH_; k0 += 32) {
    v16h a0 = load_a16(h2h, NH_, m0, k0);
    v16h a1 = load_a16(h2h, NH_, m0 + 16, k0);
#pragma unroll
    for (int c = 0; c < 4; ++c) {
      v16h b = load_b16(Bc, NH_, 16 * c, k0);
      acc[0][c] = wmma_f16(a0, b, acc[0][c]);
      acc[1][c] = wmma_f16(a1, b, acc[1][c]);
    }
  }
  int lane = threadIdx.x & 31;
  int hf = lane >> 4;
  int bb = m0 >> 10;
  size_t bh = (size_t)bb * H_ + head;
#pragma unroll
  for (int rt = 0; rt < 2; ++rt) {
    int nl = (m0 & 1023) + rt * 16 + 8 * hf;
#pragma unroll
    for (int c = 0; c < 4; ++c) {
      int d = 16 * c + (lane & 15);
      _Float16* p1 = hp + (bh * N_ + nl) * DH_ + d;   // row-major [n][d]
      _Float16* p2 = hpT + (bh * DH_ + d) * N_ + nl;  // transposed [d][n]
      v8h tv;
#pragma unroll
      for (int i = 0; i < 8; ++i) {
        _Float16 hv = (_Float16)acc[rt][c][i];
        p1[(size_t)i * DH_] = hv;
        tv[i] = hv;
      }
      *(v8h*)p2 = tv;
    }
  }
}

// ------------------------- flash-style attention ---------------------------
// Per wave: 16 queries, online softmax over 1024 keys in blocks of 32.
// Computes S^T (keys as rows) so each query's scores live in one lane pair;
// P^T is re-laid-out into the WMMA B fragment with register shuffles only.
__global__ __launch_bounds__(32) void attn_kernel(const _Float16* hp,
                                                  const _Float16* hpT,
                                                  _Float16* outc) {
  int blk = blockIdx.x;            // 16 * 4 * 64 = 4096
  int qt = blk & 63;
  int head = (blk >> 6) & 3;
  int b = blk >> 8;
  int q0 = qt * 16;
  size_t bh = (size_t)b * H_ + head;
  const _Float16* Q = hp + bh * N_ * DH_;    // [n][d]
  const _Float16* VT = hpT + bh * DH_ * N_;  // [d][n]
  int lane = threadIdx.x & 31;
  int qc = lane & 15, hf = lane >> 4;

  v16h qb0 = load_b16(Q, DH_, q0, 0);        // Q^T fragment, d = 0..31
  v16h qb1 = load_b16(Q, DH_, q0, 32);       // d = 32..63
  v8f oa0 = {}, oa1 = {}, oa2 = {}, oa3 = {};
  float mrun = -3.0e38f, lrun = 0.f;
  v8f zero = {};

  for (int kb = 0; kb < N_; kb += 32) {
    // S^T tiles: keys kb..kb+15 (t0) and kb+16..kb+31 (t1), cols = queries
    v8f s0 = wmma_f16(load_a16(Q, DH_, kb, 0), qb0, zero);
    s0 = wmma_f16(load_a16(Q, DH_, kb, 32), qb1, s0);
    v8f s1 = wmma_f16(load_a16(Q, DH_, kb + 16, 0), qb0, zero);
    s1 = wmma_f16(load_a16(Q, DH_, kb + 16, 32), qb1, s1);

    // leaky relu + per-query max (lane pair shares a query column)
    float mx = -3.0e38f;
#pragma unroll
    for (int i = 0; i < 8; ++i) {
      s0[i] = s0[i] > 0.f ? s0[i] : ALPHA_ * s0[i];
      s1[i] = s1[i] > 0.f ? s1[i] : ALPHA_ * s1[i];
      mx = fmaxf(mx, fmaxf(s0[i], s1[i]));
    }
    mx = fmaxf(mx, __shfl_xor(mx, 16, 32));
    float mnew = fmaxf(mrun, mx);
    float corr = __expf(mrun - mnew);

    float p0[8], p1[8], ls = 0.f;
#pragma unroll
    for (int i = 0; i < 8; ++i) {
      p0[i] = __expf(s0[i] - mnew);
      p1[i] = __expf(s1[i] - mnew);
      ls += p0[i] + p1[i];
    }
    ls += __shfl_xor(ls, 16, 32);
    lrun = lrun * corr + ls;
    mrun = mnew;
#pragma unroll
    for (int i = 0; i < 8; ++i) {
      oa0[i] *= corr; oa1[i] *= corr; oa2[i] *= corr; oa3[i] *= corr;
    }

    // Re-lay P^T (C layout) into B fragment via lane-pair shuffles:
    // lane half selects key group 16*hf..16*hf+15 for its query column.
    v16h pb;
#pragma unroll
    for (int j = 0; j < 8; ++j) {
      float o0v = __shfl_xor(p0[j], 16, 32);  // partner's tile0 rows 8..15
      float o1v = __shfl_xor(p1[j], 16, 32);  // partner's tile1 rows 0..7
      pb[j] = (_Float16)(hf ? o1v : p0[j]);
      pb[j + 8] = (_Float16)(hf ? p1[j] : o0v);
    }

    // out^T += V^T(block) @ P^T : rows = d (4 tiles), cols = queries
    oa0 = wmma_f16(load_a16(VT, N_, 0, kb), pb, oa0);
    oa1 = wmma_f16(load_a16(VT, N_, 16, kb), pb, oa1);
    oa2 = wmma_f16(load_a16(VT, N_, 32, kb), pb, oa2);
    oa3 = wmma_f16(load_a16(VT, N_, 48, kb), pb, oa3);
  }

  float inv = 1.0f / lrun;
  // store concat-head layout outc[(b,n)][head*64 + d], f16 for the conv GEMM
  _Float16* dst =
      outc + ((size_t)b * N_ + q0 + qc) * NO_ + head * DH_ + 8 * hf;
  v8h t0, t1, t2, t3;
#pragma unroll
  for (int i = 0; i < 8; ++i) {
    t0[i] = (_Float16)(oa0[i] * inv);
    t1[i] = (_Float16)(oa1[i] * inv);
    t2[i] = (_Float16)(oa2[i] * inv);
    t3[i] = (_Float16)(oa3[i] * inv);
  }
  *(v8h*)(dst + 0) = t0;
  *(v8h*)(dst + 16) = t1;
  *(v8h*)(dst + 32) = t2;
  *(v8h*)(dst + 48) = t3;
}

// --------------------------- conv1d (k=3) GEMM -----------------------------
// y[b][n][o] = sum_k sum_i outc[b][n+k-1][i] * w[o][i][k] + conv_b[o]
// 32x64 per wave: grid = (16384/32) * (256/64) = 2048
__global__ __launch_bounds__(32) void conv_kernel(const _Float16* outc,
                                                  const _Float16* convwT,
                                                  const float* conv_b,
                                                  float* y) {
  int tile = blockIdx.x;
  int og = tile & 3, rg = tile >> 2;
  int r0 = rg * 32, o0 = og * 64;
  int b = r0 >> 10, n0 = r0 & 1023;
  int lane = threadIdx.x & 31;
  int rr = lane & 15, hf = lane >> 4;
  v8h z = {};
  v8f acc[2][4] = {};
#pragma unroll
  for (int k = 0; k < 3; ++k) {
    const _Float16* Bc = convwT + (size_t)k * NO_ * NO_;
    int nrow0 = n0 + rr + k - 1;
    int nrow1 = n0 + 16 + rr + k - 1;
    bool ok0 = (nrow0 >= 0) && (nrow0 < N_);
    bool ok1 = (nrow1 >= 0) && (nrow1 < N_);
    const _Float16* Ar0 = outc + ((size_t)b * N_ + (ok0 ? nrow0 : 0)) * NO_;
    const _Float16* Ar1 = outc + ((size_t)b * N_ + (ok1 ? nrow1 : 0)) * NO_;
    for (int k0 = 0; k0 < NO_; k0 += 32) {
      const _Float16* p0 = Ar0 + k0 + 16 * hf;
      const _Float16* p1 = Ar1 + k0 + 16 * hf;
      v8h lo0 = ok0 ? *(const v8h*)p0 : z;
      v8h hi0 = ok0 ? *(const v8h*)(p0 + 8) : z;
      v8h lo1 = ok1 ? *(const v8h*)p1 : z;
      v8h hi1 = ok1 ? *(const v8h*)(p1 + 8) : z;
      v16h a0, a1;
#pragma unroll
      for (int i = 0; i < 8; ++i) {
        a0[i] = lo0[i]; a0[i + 8] = hi0[i];
        a1[i] = lo1[i]; a1[i + 8] = hi1[i];
      }
#pragma unroll
      for (int c = 0; c < 4; ++c) {
        v16h bm = load_b16(Bc, NO_, o0 + 16 * c, k0);
        acc[0][c] = wmma_f16(a0, bm, acc[0][c]);
        acc[1][c] = wmma_f16(a1, bm, acc[1][c]);
      }
    }
  }
#pragma unroll
  for (int c = 0; c < 4; ++c) {
    int o = o0 + 16 * c + (lane & 15);
    float bias = conv_b[o];
#pragma unroll
    for (int rt = 0; rt < 2; ++rt) {
      float* dst = y + ((size_t)(r0 + rt * 16 + 8 * hf)) * NO_ + o;
#pragma unroll
      for (int i = 0; i < 8; ++i)
        dst[(size_t)i * NO_] = acc[rt][c][i] + bias;
    }
  }
}

// BN2 apply + relu -> final f32 output (B,N,NOUT)
__global__ void bn2_apply_kernel(const float* y, const float* scale,
                                 const float* shift, float* out) {
  size_t i = (size_t)blockIdx.x * 256 + threadIdx.x;
  int c = (int)(i & 255);
  out[i] = fmaxf(y[i] * scale[c] + shift[c], 0.f);
}

// ------------------------------- launcher ----------------------------------

extern "C" void kernel_launch(void* const* d_in, const int* in_sizes, int n_in,
                              void* d_out, int out_size, void* d_ws,
                              size_t ws_size, hipStream_t stream) {
  const float* x = (const float*)d_in[0];          // (16,1024,128)
  const float* adj = (const float*)d_in[1];        // (1024,1024)
  const float* gc_w = (const float*)d_in[2];       // (128,256)
  const float* gc_b = (const float*)d_in[3];       // (256,)
  const float* bn1_g = (const float*)d_in[4];
  const float* bn1_b = (const float*)d_in[5];
  const float* gat_w = (const float*)d_in[6];      // (4,256,64)
  const float* conv_w = (const float*)d_in[7];     // (256,256,3)
  const float* conv_b = (const float*)d_in[8];
  const float* bn2_g = (const float*)d_in[9];
  const float* bn2_b = (const float*)d_in[10];
  float* out = (float*)d_out;                      // (16,1024,256)

  char* w = (char*)d_ws;
  _Float16* xh = (_Float16*)w;          w += (size_t)16384 * 128 * 2;   // 4 MB
  _Float16* gcwT = (_Float16*)w;        w += (size_t)256 * 128 * 2;
  _Float16* adjh = (_Float16*)w;        w += (size_t)1024 * 1024 * 2;   // 2 MB
  _Float16* h0t = (_Float16*)w;         w += (size_t)16 * 256 * 1024 * 2;
  float* h1 = (float*)w;                w += (size_t)16384 * 256 * 4;   // 16 MB
  float* bn1_scale = (float*)w;         w += 1024;
  float* bn1_shift = (float*)w;         w += 1024;
  _Float16* h2h = (_Float16*)w;         w += (size_t)16384 * 256 * 2;
  _Float16* gatwT = (_Float16*)w;       w += (size_t)4 * 64 * 256 * 2;
  _Float16* hp = (_Float16*)w;          w += (size_t)16 * 4 * 1024 * 64 * 2;
  _Float16* hpT = (_Float16*)w;         w += (size_t)16 * 4 * 1024 * 64 * 2;
  _Float16* outc = (_Float16*)w;        w += (size_t)16384 * 256 * 2;
  _Float16* convwT = (_Float16*)w;      w += (size_t)3 * 256 * 256 * 2;
  float* y = (float*)w;                 w += (size_t)16384 * 256 * 4;   // 16 MB
  float* bn2_scale = (float*)w;         w += 1024;
  float* bn2_shift = (float*)w;         w += 1024;

  // -- pack f16 operands
  pack_f32_to_f16<<<(16384 * 128) / 256, 256, 0, stream>>>(x, xh, 16384 * 128);
  pack_gcw<<<(256 * 128) / 256, 256, 0, stream>>>(gc_w, gcwT);
  pack_f32_to_f16<<<(1024 * 1024) / 256, 256, 0, stream>>>(adj, adjh,
                                                           1024 * 1024);
  pack_gatw<<<(4 * 64 * 256) / 256, 256, 0, stream>>>(gat_w, gatwT);
  pack_convw<<<(3 * 256 * 256) / 256, 256, 0, stream>>>(conv_w, convwT);

  // -- GCN: x @ W, then adj @ h + b, relu
  gemm1_kernel<<<2048, 32, 0, stream>>>(xh, gcwT, h0t);
  gemm2_kernel<<<2048, 32, 0, stream>>>(adjh, h0t, gc_b, h1);

  // -- BN1 + relu -> f16
  bn_stats_kernel<<<256, 256, 0, stream>>>(h1, 16384, 256, bn1_g, bn1_b,
                                           bn1_scale, bn1_shift);
  bn1_apply_kernel<<<(16384 * 256) / 256, 256, 0, stream>>>(h1, bn1_scale,
                                                            bn1_shift, h2h);

  // -- GAT projection + attention
  gemm3_kernel<<<2048, 32, 0, stream>>>(h2h, gatwT, hp, hpT);
  attn_kernel<<<4096, 32, 0, stream>>>(hp, hpT, outc);

  // -- TCN conv + BN2 + relu
  conv_kernel<<<2048, 32, 0, stream>>>(outc, convwT, conv_b, y);
  bn_stats_kernel<<<256, 256, 0, stream>>>(y, 16384, 256, bn2_g, bn2_b,
                                           bn2_scale, bn2_shift);
  bn2_apply_kernel<<<(16384 * 256) / 256, 256, 0, stream>>>(y, bn2_scale,
                                                            bn2_shift, out);
}
